// GRU_54614804135975
// MI455X (gfx1250) — matrix operand order repaired and checked by reading, not compiled, not run on hardware
//
#include <hip/hip_runtime.h>
#include <hip/hip_bf16.h>

// ---------------------------------------------------------------------------
// GRU (B=1024, D=256, H=1024, T=128) + per-frame BatchNorm, bf16 WMMA path.
// ---------------------------------------------------------------------------

#define B_ 1024
#define D_ 256
#define H_ 1024
#define T_ 128

typedef __attribute__((ext_vector_type(16))) __bf16 v16bf;
typedef __attribute__((ext_vector_type(8)))  float  v8f;
typedef __attribute__((ext_vector_type(4)))  unsigned int u32x4;

struct FragBits { u32x4 lo, hi; };

__device__ __forceinline__ unsigned short f2bf(float f) {
    unsigned int u = __builtin_bit_cast(unsigned int, f);
    unsigned int lsb = (u >> 16) & 1u;
    u += 0x7fffu + lsb;                 // round-to-nearest-even
    return (unsigned short)(u >> 16);
}

// A-matrix fragment, 16x32 bf16 tile at (row0, k0), row-major src, leading dim ld.
// lane L: m = L&15 ; halves 0-7 -> k = k0 + ((L>>4)*8) ; halves 8-15 -> +16.
__device__ __forceinline__ v16bf load_frag_a(const unsigned short* __restrict__ base,
                                             int ld, int row0, int k0, int lane) {
    int m  = lane & 15;
    int ks = (lane >> 4) << 3;          // 0 or 8
    const unsigned short* p = base + (size_t)(row0 + m) * ld + k0 + ks;
    FragBits f;
    f.lo = *reinterpret_cast<const u32x4*>(p);        // k .. k+7
    f.hi = *reinterpret_cast<const u32x4*>(p + 16);   // k+16 .. k+23
    return __builtin_bit_cast(v16bf, f);
}

// B-matrix fragment, 32x16 (KxN) where element (k,n) = W[n][k], W row-major (N,K).
// lane L: n = L&15 ; halves 0-7 -> k = k0 + ((L>>4)*8) ; halves 8-15 -> +16.
__device__ __forceinline__ v16bf load_frag_b(const unsigned short* __restrict__ W,
                                             int ld, int k0, int n0, int lane) {
    int n  = lane & 15;
    int ko = (lane >> 4) << 3;          // 0 or 8
    const unsigned short* p = W + (size_t)(n0 + n) * ld + k0 + ko;
    FragBits f;
    f.lo = *reinterpret_cast<const u32x4*>(p);
    f.hi = *reinterpret_cast<const u32x4*>(p + 16);
    return __builtin_bit_cast(v16bf, f);
}

__device__ __forceinline__ v8f wmma_bf16(v16bf a, v16bf b, v8f c) {
    return __builtin_amdgcn_wmma_f32_16x16x32_bf16(false, a, false, b,
                                                   (short)0, c, false, false);
}

__device__ __forceinline__ float sigmoidf(float x) {
    return 1.0f / (1.0f + __expf(-x));
}

// ---------------------------------------------------------------------------
// Conversion / init kernels
// ---------------------------------------------------------------------------
__global__ void f32_to_bf16_k(const float* __restrict__ in,
                              unsigned short* __restrict__ out, int n) {
    int i = blockIdx.x * blockDim.x + threadIdx.x;
    if (i < n) out[i] = f2bf(in[i]);
}

__global__ void init_h_k(const float* __restrict__ h0, float* __restrict__ hf,
                         unsigned short* __restrict__ hb, int n) {
    int i = blockIdx.x * blockDim.x + threadIdx.x;
    if (i < n) { float v = h0[i]; hf[i] = v; hb[i] = f2bf(v); }
}

// ---------------------------------------------------------------------------
// Gates kernel: r = sig(gi_r+gh_r), z = sig(gi_z+gh_z),
//               n = tanh(gi_n + r*gh_n), h' = (1-z)*n + z*h
// Wave tile 32x32 of H; block 8 waves = 128 rows x 64 cols.
// grid = (H/64, B/128)
// ---------------------------------------------------------------------------
__global__ __launch_bounds__(256) void gru_gates(
    const unsigned short* __restrict__ x_bf,   // B x D
    const unsigned short* __restrict__ h_bf,   // B x H
    const float*          __restrict__ h_f32,  // B x H
    const unsigned short* __restrict__ Wih,    // 3H x D
    const unsigned short* __restrict__ Whh,    // 3H x H
    const float* __restrict__ b_ih, const float* __restrict__ b_hh,
    float* __restrict__ hn_f32, unsigned short* __restrict__ hn_bf) {

    const int lane = threadIdx.x & 31;
    const int wave = threadIdx.x >> 5;
    const int row0 = blockIdx.y * 128 + (wave >> 1) * 32;
    const int col0 = blockIdx.x * 64  + (wave & 1)  * 32;

    v8f accR[2][2], accZ[2][2], accN[2][2], accH[2][2];
    const v8f vz = {0.f,0.f,0.f,0.f,0.f,0.f,0.f,0.f};
#pragma unroll
    for (int i = 0; i < 2; ++i)
#pragma unroll
        for (int j = 0; j < 2; ++j) {
            accR[i][j] = vz; accZ[i][j] = vz; accN[i][j] = vz; accH[i][j] = vz;
        }

    // ---- phase 1: gi = x @ W_ih^T -------------------------------------
    for (int k = 0; k < D_; k += 32) {
        v16bf a0 = load_frag_a(x_bf, D_, row0,      k, lane);
        v16bf a1 = load_frag_a(x_bf, D_, row0 + 16, k, lane);
#pragma unroll
        for (int j = 0; j < 2; ++j) {
            int n0 = col0 + j * 16;
            v16bf br = load_frag_b(Wih, D_, k, n0,          lane);
            v16bf bz = load_frag_b(Wih, D_, k, n0 + H_,     lane);
            v16bf bn = load_frag_b(Wih, D_, k, n0 + 2 * H_, lane);
            accR[0][j] = wmma_bf16(a0, br, accR[0][j]);
            accR[1][j] = wmma_bf16(a1, br, accR[1][j]);
            accZ[0][j] = wmma_bf16(a0, bz, accZ[0][j]);
            accZ[1][j] = wmma_bf16(a1, bz, accZ[1][j]);
            accN[0][j] = wmma_bf16(a0, bn, accN[0][j]);
            accN[1][j] = wmma_bf16(a1, bn, accN[1][j]);
        }
    }
    // ---- phase 2: gh = h @ W_hh^T (r,z fused; n-part kept separate) ----
    for (int k = 0; k < H_; k += 32) {
        v16bf a0 = load_frag_a(h_bf, H_, row0,      k, lane);
        v16bf a1 = load_frag_a(h_bf, H_, row0 + 16, k, lane);
#pragma unroll
        for (int j = 0; j < 2; ++j) {
            int n0 = col0 + j * 16;
            v16bf br = load_frag_b(Whh, H_, k, n0,          lane);
            v16bf bz = load_frag_b(Whh, H_, k, n0 + H_,     lane);
            v16bf bn = load_frag_b(Whh, H_, k, n0 + 2 * H_, lane);
            accR[0][j] = wmma_bf16(a0, br, accR[0][j]);
            accR[1][j] = wmma_bf16(a1, br, accR[1][j]);
            accZ[0][j] = wmma_bf16(a0, bz, accZ[0][j]);
            accZ[1][j] = wmma_bf16(a1, bz, accZ[1][j]);
            accH[0][j] = wmma_bf16(a0, bn, accH[0][j]);
            accH[1][j] = wmma_bf16(a1, bn, accH[1][j]);
        }
    }

    // ---- epilogue: gates + state update --------------------------------
    const int mo = (lane >> 4) << 3;     // 0 or 8
    const int nl = lane & 15;
#pragma unroll
    for (int i = 0; i < 2; ++i)
#pragma unroll
        for (int j = 0; j < 2; ++j) {
            int n = col0 + j * 16 + nl;
            float bir = b_ih[n]          + b_hh[n];
            float biz = b_ih[H_ + n]     + b_hh[H_ + n];
            float bin = b_ih[2 * H_ + n];
            float bhn = b_hh[2 * H_ + n];
#pragma unroll
            for (int e = 0; e < 8; ++e) {
                int m = row0 + i * 16 + mo + e;
                float r  = sigmoidf(accR[i][j][e] + bir);
                float z  = sigmoidf(accZ[i][j][e] + biz);
                float nn = tanhf(accN[i][j][e] + bin + r * (accH[i][j][e] + bhn));
                float hp = h_f32[(size_t)m * H_ + n];
                float hv = (1.0f - z) * nn + z * hp;
                hn_f32[(size_t)m * H_ + n] = hv;
                hn_bf [(size_t)m * H_ + n] = f2bf(hv);
            }
        }
}

// ---------------------------------------------------------------------------
// Output projection: out = h' @ W_lin^T + b_lin ; also emits bf16 x for t+1.
// grid = (D/64, B/128)
// ---------------------------------------------------------------------------
__global__ __launch_bounds__(256) void gru_out(
    const unsigned short* __restrict__ h_bf,   // B x H
    const unsigned short* __restrict__ Wlin,   // D x H
    const float* __restrict__ b_lin,
    float* __restrict__ out,                   // B x D (frame slice of d_out)
    unsigned short* __restrict__ x_next) {     // B x D

    const int lane = threadIdx.x & 31;
    const int wave = threadIdx.x >> 5;
    const int row0 = blockIdx.y * 128 + (wave >> 1) * 32;
    const int col0 = blockIdx.x * 64  + (wave & 1)  * 32;

    v8f acc[2][2];
    const v8f vz = {0.f,0.f,0.f,0.f,0.f,0.f,0.f,0.f};
#pragma unroll
    for (int i = 0; i < 2; ++i)
#pragma unroll
        for (int j = 0; j < 2; ++j) acc[i][j] = vz;

    for (int k = 0; k < H_; k += 32) {
        v16bf a0 = load_frag_a(h_bf, H_, row0,      k, lane);
        v16bf a1 = load_frag_a(h_bf, H_, row0 + 16, k, lane);
#pragma unroll
        for (int j = 0; j < 2; ++j) {
            v16bf b = load_frag_b(Wlin, H_, k, col0 + j * 16, lane);
            acc[0][j] = wmma_bf16(a0, b, acc[0][j]);
            acc[1][j] = wmma_bf16(a1, b, acc[1][j]);
        }
    }

    const int mo = (lane >> 4) << 3;
    const int nl = lane & 15;
#pragma unroll
    for (int i = 0; i < 2; ++i)
#pragma unroll
        for (int j = 0; j < 2; ++j) {
            int n = col0 + j * 16 + nl;
            float bl = b_lin[n];
#pragma unroll
            for (int e = 0; e < 8; ++e) {
                int m = row0 + i * 16 + mo + e;
                float v = acc[i][j][e] + bl;
                out[(size_t)m * D_ + n]    = v;
                x_next[(size_t)m * D_ + n] = f2bf(v);
            }
        }
}

// ---------------------------------------------------------------------------
// BatchNorm1d(affine=False), training stats per (t, d) over batch. In place.
// block = 256 threads = 64 features x 4 batch-strips; grid = T * (D/64).
// ---------------------------------------------------------------------------
__global__ __launch_bounds__(256) void bn_norm(float* __restrict__ outs) {
    const int t  = blockIdx.x >> 2;
    const int g  = blockIdx.x & 3;
    const int dl = threadIdx.x & 63;
    const int bl = threadIdx.x >> 6;
    float* base = outs + (size_t)t * B_ * D_ + g * 64 + dl;

    float s = 0.f, ss = 0.f;
    for (int b = bl; b < B_; b += 4) {
        float v = base[(size_t)b * D_];
        s += v; ss += v * v;
    }
    __shared__ float rs[4][64], rss[4][64];
    __shared__ float mean_s[64], inv_s[64];
    rs[bl][dl] = s; rss[bl][dl] = ss;
    __syncthreads();
    if (bl == 0) {
        float S  = rs[0][dl]  + rs[1][dl]  + rs[2][dl]  + rs[3][dl];
        float SS = rss[0][dl] + rss[1][dl] + rss[2][dl] + rss[3][dl];
        float mean = S * (1.0f / B_);
        float var  = SS * (1.0f / B_) - mean * mean;
        mean_s[dl] = mean;
        inv_s[dl]  = rsqrtf(var + 1e-5f);
    }
    __syncthreads();
    float mean = mean_s[dl], inv = inv_s[dl];
    for (int b = bl; b < B_; b += 4) {
        base[(size_t)b * D_] = (base[(size_t)b * D_] - mean) * inv;
    }
}

// ---------------------------------------------------------------------------
// Host-side launcher
// ---------------------------------------------------------------------------
extern "C" void kernel_launch(void* const* d_in, const int* in_sizes, int n_in,
                              void* d_out, int out_size, void* d_ws, size_t ws_size,
                              hipStream_t stream) {
    const float* x     = (const float*)d_in[0];
    const float* h0    = (const float*)d_in[1];
    const float* W_ih  = (const float*)d_in[2];
    const float* W_hh  = (const float*)d_in[3];
    const float* b_ih  = (const float*)d_in[4];
    const float* b_hh  = (const float*)d_in[5];
    const float* W_lin = (const float*)d_in[6];
    const float* b_lin = (const float*)d_in[7];
    (void)in_sizes; (void)n_in; (void)out_size; (void)ws_size;

    // workspace carve-out (256B aligned slices)
    char* ws = (char*)d_ws;
    auto carve = [&](size_t bytes) {
        char* p = ws;
        ws += (bytes + 255) & ~(size_t)255;
        return p;
    };
    unsigned short* Wih_bf  = (unsigned short*)carve((size_t)3 * H_ * D_ * 2);
    unsigned short* Whh_bf  = (unsigned short*)carve((size_t)3 * H_ * H_ * 2);
    unsigned short* Wlin_bf = (unsigned short*)carve((size_t)D_ * H_ * 2);
    unsigned short* x_bf    = (unsigned short*)carve((size_t)B_ * D_ * 2);
    float*          hf0     = (float*)carve((size_t)B_ * H_ * 4);
    float*          hf1     = (float*)carve((size_t)B_ * H_ * 4);
    unsigned short* hb0     = (unsigned short*)carve((size_t)B_ * H_ * 2);
    unsigned short* hb1     = (unsigned short*)carve((size_t)B_ * H_ * 2);
    float* hf[2] = {hf0, hf1};
    unsigned short* hb[2] = {hb0, hb1};

    // one-time conversions
    {
        int n;
        n = 3 * H_ * D_;
        f32_to_bf16_k<<<(n + 255) / 256, 256, 0, stream>>>(W_ih, Wih_bf, n);
        n = 3 * H_ * H_;
        f32_to_bf16_k<<<(n + 255) / 256, 256, 0, stream>>>(W_hh, Whh_bf, n);
        n = D_ * H_;
        f32_to_bf16_k<<<(n + 255) / 256, 256, 0, stream>>>(W_lin, Wlin_bf, n);
        n = B_ * D_;
        f32_to_bf16_k<<<(n + 255) / 256, 256, 0, stream>>>(x, x_bf, n);
        n = B_ * H_;
        init_h_k<<<(n + 255) / 256, 256, 0, stream>>>(h0, hf[0], hb[0], n);
    }

    // sequential recurrence: stream ordering is the inter-step barrier
    int cur = 0;
    for (int t = 0; t < T_; ++t) {
        int nxt = cur ^ 1;
        dim3 gA(H_ / 64, B_ / 128);
        gru_gates<<<gA, 256, 0, stream>>>(x_bf, hb[cur], hf[cur], Wih_bf, Whh_bf,
                                          b_ih, b_hh, hf[nxt], hb[nxt]);
        dim3 gB(D_ / 64, B_ / 128);
        gru_out<<<gB, 256, 0, stream>>>(hb[nxt], Wlin_bf, b_lin,
                                        (float*)d_out + (size_t)t * B_ * D_, x_bf);
        cur = nxt;
    }

    // per-frame batch normalization, in place on d_out
    bn_norm<<<T_ * (D_ / 64), 256, 0, stream>>>((float*)d_out);
}
